// MultiHeadedAttention_1881195676077
// MI455X (gfx1250) — compile-verified, matrix operand order checked by reading, not run on hardware
//
#include <hip/hip_runtime.h>

// Problem constants (match reference)
#define Bc   2
#define Tc   2048
#define Dc   1024
#define Hc   16
#define DKc  64
#define LN_EPS 1e-5f

typedef __attribute__((ext_vector_type(16))) _Float16 v16h;
typedef __attribute__((ext_vector_type(8)))  float    v8f;

union HFrag {
    v16h v;
    unsigned u[8];
};

// Build a WMMA fragment from one LDS "row" (row-major, K contiguous).
// Per CDNA5 ISA 16-bit A layout: lanes 0-15 take K {0..7,16..23}, lanes 16-31 take
// K {8..15,24..31}; each VGPR holds a consecutive K pair -> aligned u32 LDS reads.
__device__ __forceinline__ void load_frag(HFrag& f, const _Float16* rowptr, int kb) {
#pragma unroll
    for (int p = 0; p < 8; ++p) {
        const int ki = ((p >> 2) << 4) + kb + ((p & 3) << 1);
        f.u[p] = *(const unsigned*)(rowptr + ki);
    }
}

// CDNA5 async global->LDS copy (16 bytes per lane), tracked by ASYNCcnt.
// VDST holds the LDS byte address (= low 32 bits of the generic shared pointer,
// per the ISA aperture mapping); VADDR is the 64-bit global address.
__device__ __forceinline__ void async_cp16(void* ldsp, const void* gp) {
    asm volatile("global_load_async_to_lds_b128 %0, %1, off"
                 :: "v"((unsigned)(unsigned long long)ldsp), "v"(gp)
                 : "memory");
}
__device__ __forceinline__ void async_wait0() {
    asm volatile("s_wait_asynccnt 0" ::: "memory");
}

// ---------------------------------------------------------------------------
// C[M,N] = A[M,K] @ W[N,K]^T + bias (+ residual), WMMA f16 -> f32
// block = 256 thr (8 waves), tile 128(M) x 64(N), BK = 32
// ---------------------------------------------------------------------------
template <typename AT, typename OT, bool RES>
__global__ __launch_bounds__(256) void gemm_bias_kernel(
    const AT* __restrict__ A, const float* __restrict__ W,
    const float* __restrict__ bias, const float* __restrict__ residual,
    OT* __restrict__ C, int M, int N, int K)
{
    constexpr int BK = 32;
    __shared__ __align__(16) _Float16 As[128 * BK];
    __shared__ __align__(16) _Float16 Bs[64 * BK];

    const int tid  = threadIdx.x;
    const int lane = tid & 31;          // wave32
    const int wv   = tid >> 5;          // 8 waves
    const int row0 = blockIdx.y * 128;
    const int col0 = blockIdx.x * 64;

    v8f acc[4] = {};

    const int lr = tid >> 3;            // 0..31
    const int lc = (tid & 7) * 4;       // 0..28 step 4

    for (int k0 = 0; k0 < K; k0 += BK) {
        // stage A tile 128x32 (cvt to f16)
#pragma unroll
        for (int p = 0; p < 4; ++p) {
            const int r = p * 32 + lr;
            const AT* src = A + (size_t)(row0 + r) * K + (k0 + lc);
            _Float16* dst = &As[r * BK + lc];
#pragma unroll
            for (int j = 0; j < 4; ++j) dst[j] = (_Float16)(float)src[j];
        }
        // stage W tile 64x32 (row n of W == column n of B)
#pragma unroll
        for (int p = 0; p < 2; ++p) {
            const int r = p * 32 + lr;
            const float* src = W + (size_t)(col0 + r) * K + (k0 + lc);
            _Float16* dst = &Bs[r * BK + lc];
#pragma unroll
            for (int j = 0; j < 4; ++j) dst[j] = (_Float16)src[j];
        }
        if (k0 + BK < K)
            __builtin_prefetch(A + (size_t)(row0 + lr) * K + (k0 + BK + lc), 0, 0);
        __syncthreads();

        const int kb = (lane >> 4) << 3;
        HFrag af, bf;
        load_frag(af, &As[(wv * 16 + (lane & 15)) * BK], kb);
#pragma unroll
        for (int nt = 0; nt < 4; ++nt) {
            load_frag(bf, &Bs[(nt * 16 + (lane & 15)) * BK], kb);
            acc[nt] = __builtin_amdgcn_wmma_f32_16x16x32_f16(
                false, af.v, false, bf.v, (short)0, acc[nt], false, false);
        }
        __syncthreads();
    }

    const int mbase = row0 + wv * 16 + ((lane >> 4) << 3);
    const int nlo = lane & 15;
#pragma unroll
    for (int nt = 0; nt < 4; ++nt) {
        const int col = col0 + nt * 16 + nlo;
        const float bval = bias[col];
#pragma unroll
        for (int r = 0; r < 8; ++r) {
            const int row = mbase + r;
            float v = acc[nt][r] + bval;
            if (RES) v += residual[(size_t)row * N + col];
            C[(size_t)row * N + col] = (OT)v;
        }
    }
}

// ---------------------------------------------------------------------------
// Scores: S[b,h,q,k] = (q . k) / sqrt(DK); q/k f16 in [B,T,D], head slice.
// Q/K tiles staged via GLOBAL_LOAD_ASYNC_TO_LDS_B128 (ASYNCcnt path).
// Raw scores streamed to d_out attn region with non-temporal stores.
// ---------------------------------------------------------------------------
__global__ __launch_bounds__(256) void scores_kernel(
    const _Float16* __restrict__ qh, const _Float16* __restrict__ kh,
    float* __restrict__ Sout)
{
    __shared__ __align__(16) _Float16 Qs[128 * 32];
    __shared__ __align__(16) _Float16 Ks[64 * 32];

    const int tid  = threadIdx.x;
    const int lane = tid & 31;
    const int wv   = tid >> 5;
    const int bh   = blockIdx.z;
    const int b    = bh >> 4;           // Hc == 16
    const int h    = bh & 15;
    const _Float16* Q  = qh + (size_t)b * Tc * Dc + h * DKc;
    const _Float16* Kp = kh + (size_t)b * Tc * Dc + h * DKc;
    const int q0 = blockIdx.y * 128;
    const int c0 = blockIdx.x * 64;

    v8f acc[4] = {};

    for (int k0 = 0; k0 < DKc; k0 += 32) {
        // Q tile: 128 rows x 64B = 512 x 16B chunks, 2 per thread (async)
#pragma unroll
        for (int c = 0; c < 2; ++c) {
            const int chunk = tid + c * 256;
            const int row   = chunk >> 2;
            const int col8  = (chunk & 3) * 8;   // f16 elements
            async_cp16(&Qs[row * 32 + col8],
                       Q + (size_t)(q0 + row) * Dc + k0 + col8);
        }
        // K tile: 64 rows x 64B = 256 x 16B chunks, 1 per thread (async)
        {
            const int row  = tid >> 2;
            const int col8 = (tid & 3) * 8;
            async_cp16(&Ks[row * 32 + col8],
                       Kp + (size_t)(c0 + row) * Dc + k0 + col8);
        }
        async_wait0();
        __syncthreads();

        const int kb = (lane >> 4) << 3;
        HFrag af, bf;
        load_frag(af, &Qs[(wv * 16 + (lane & 15)) * 32], kb);
#pragma unroll
        for (int nt = 0; nt < 4; ++nt) {
            load_frag(bf, &Ks[(nt * 16 + (lane & 15)) * 32], kb);
            acc[nt] = __builtin_amdgcn_wmma_f32_16x16x32_f16(
                false, af.v, false, bf.v, (short)0, acc[nt], false, false);
        }
        __syncthreads();
    }

    const int mbase = q0 + wv * 16 + ((lane >> 4) << 3);
    const int nlo = lane & 15;
#pragma unroll
    for (int nt = 0; nt < 4; ++nt) {
        const int col = c0 + nt * 16 + nlo;
#pragma unroll
        for (int r = 0; r < 8; ++r) {
            const int row = mbase + r;
            __builtin_nontemporal_store(acc[nt][r] * 0.125f,
                                        &Sout[((size_t)bh * Tc + row) * Tc + col]);
        }
    }
}

// ---------------------------------------------------------------------------
// softmax_one stats only: rowmax = max(x); rowinv = 1/(1 + sum(exp(x-max)))
// one block per (b,h,q) row of T=2048. Streaming non-temporal reads.
// ---------------------------------------------------------------------------
__global__ __launch_bounds__(256) void softmax_stats_kernel(
    const float* __restrict__ attn, float* __restrict__ rowmax,
    float* __restrict__ rowinv)
{
    __shared__ float red[256];
    const int tid = threadIdx.x;
    const float* p = attn + (size_t)blockIdx.x * Tc;

    float vals[8];
    float mx = -3.4e38f;
#pragma unroll
    for (int i = 0; i < 8; ++i) {
        vals[i] = __builtin_nontemporal_load(p + tid + 256 * i);
        mx = fmaxf(mx, vals[i]);
    }
    red[tid] = mx;
    __syncthreads();
    for (int s = 128; s > 0; s >>= 1) {
        if (tid < s) red[tid] = fmaxf(red[tid], red[tid + s]);
        __syncthreads();
    }
    mx = red[0];
    __syncthreads();

    float sum = 0.f;
#pragma unroll
    for (int i = 0; i < 8; ++i) sum += __expf(vals[i] - mx);
    red[tid] = sum;
    __syncthreads();
    for (int s = 128; s > 0; s >>= 1) {
        if (tid < s) red[tid] += red[tid + s];
        __syncthreads();
    }
    if (tid == 0) {
        rowmax[blockIdx.x] = mx;
        rowinv[blockIdx.x] = 1.f / (1.f + red[0]);
    }
}

// ---------------------------------------------------------------------------
// Fused normalize + attn@V:
//   p = exp(s - rowmax) * rowinv  -> written back to attn (final output)
//   x[q,d] = sum_k p[q,k] * v[k,d]   (WMMA, V transposed in padded LDS)
// Each attn element is touched exactly once here, so the in-place normalized
// write is deterministic. All attn traffic is non-temporal (537MB >> L2).
// ---------------------------------------------------------------------------
__global__ __launch_bounds__(256) void attnv_fused_kernel(
    float* __restrict__ attn, const _Float16* __restrict__ vh,
    const float* __restrict__ rowmax, const float* __restrict__ rowinv,
    _Float16* __restrict__ xh)
{
    __shared__ __align__(16) _Float16 As[128 * 32];
    __shared__ __align__(16) _Float16 Vs[64 * 34];  // [d][k], pad 34

    const int tid  = threadIdx.x;
    const int lane = tid & 31;
    const int wv   = tid >> 5;
    const int bh   = blockIdx.y;
    const int b    = bh >> 4;
    const int h    = bh & 15;
    float* Ap = attn + (size_t)bh * Tc * Tc;
    const _Float16* V = vh + (size_t)b * Tc * Dc + h * DKc;
    const int q0 = blockIdx.x * 128;

    v8f acc[4] = {};
    const int lr = tid >> 3;
    const int lc = (tid & 7) * 4;
    const int vk = tid >> 3;            // key row within BK chunk
    const int vd = (tid & 7) * 8;       // 8 consecutive d per thread

    // per-thread rows are fixed across the K loop -> hoist softmax stats
    float mxr[4], ivr[4];
#pragma unroll
    for (int p = 0; p < 4; ++p) {
        const size_t ridx = (size_t)bh * Tc + (q0 + p * 32 + lr);
        mxr[p] = rowmax[ridx];
        ivr[p] = rowinv[ridx];
    }

    for (int k0 = 0; k0 < Tc; k0 += 32) {
        // stage attn tile 128x32: normalize, write back, cvt f16 to LDS
#pragma unroll
        for (int p = 0; p < 4; ++p) {
            const int r = p * 32 + lr;
            float* src = Ap + (size_t)(q0 + r) * Tc + (k0 + lc);
            _Float16* dst = &As[r * 32 + lc];
#pragma unroll
            for (int j = 0; j < 4; ++j) {
                const float s = __builtin_nontemporal_load(src + j);
                const float pr = __expf(s - mxr[p]) * ivr[p];
                __builtin_nontemporal_store(pr, src + j);
                dst[j] = (_Float16)pr;
            }
        }
        // stage V chunk transposed: Vs[d][k] = v[k0+k][d]
        const _Float16* vsrc = V + (size_t)(k0 + vk) * Dc + vd;
#pragma unroll
        for (int j = 0; j < 8; ++j) Vs[(vd + j) * 34 + vk] = vsrc[j];
        __syncthreads();

        const int kb = (lane >> 4) << 3;
        HFrag af, bf;
        load_frag(af, &As[(wv * 16 + (lane & 15)) * 32], kb);
#pragma unroll
        for (int nt = 0; nt < 4; ++nt) {
            load_frag(bf, &Vs[(nt * 16 + (lane & 15)) * 34], kb);
            acc[nt] = __builtin_amdgcn_wmma_f32_16x16x32_f16(
                false, af.v, false, bf.v, (short)0, acc[nt], false, false);
        }
        __syncthreads();
    }

    const int mbase = q0 + wv * 16 + ((lane >> 4) << 3);
    const int nlo = lane & 15;
#pragma unroll
    for (int nt = 0; nt < 4; ++nt) {
        const int col = nt * 16 + nlo;  // < 64 == DKc
#pragma unroll
        for (int r = 0; r < 8; ++r) {
            const int row = mbase + r;
            xh[(size_t)b * Tc * Dc + (size_t)row * Dc + h * DKc + col] =
                (_Float16)acc[nt][r];
        }
    }
}

// ---------------------------------------------------------------------------
// Row LayerNorm over D=1024: out = (y - mu)*rsqrt(var+eps)*gamma + beta
// ---------------------------------------------------------------------------
__global__ __launch_bounds__(256) void layernorm_kernel(
    const float* __restrict__ ybuf, const float* __restrict__ gamma,
    const float* __restrict__ beta, float* __restrict__ out)
{
    __shared__ float red[256];
    const int tid = threadIdx.x;
    const float* p = ybuf + (size_t)blockIdx.x * Dc;

    float v[4];
    float s = 0.f;
#pragma unroll
    for (int i = 0; i < 4; ++i) {
        v[i] = p[tid + 256 * i];
        s += v[i];
    }
    red[tid] = s;
    __syncthreads();
    for (int st = 128; st > 0; st >>= 1) {
        if (tid < st) red[tid] += red[tid + st];
        __syncthreads();
    }
    const float mu = red[0] * (1.f / Dc);
    __syncthreads();

    float s2 = 0.f;
#pragma unroll
    for (int i = 0; i < 4; ++i) {
        const float d = v[i] - mu;
        s2 += d * d;
    }
    red[tid] = s2;
    __syncthreads();
    for (int st = 128; st > 0; st >>= 1) {
        if (tid < st) red[tid] += red[tid + st];
        __syncthreads();
    }
    const float rs = rsqrtf(red[0] * (1.f / Dc) + LN_EPS);

    float* o = out + (size_t)blockIdx.x * Dc;
#pragma unroll
    for (int i = 0; i < 4; ++i) {
        const int c = tid + 256 * i;
        o[c] = (v[i] - mu) * rs * gamma[c] + beta[c];
    }
}

// ---------------------------------------------------------------------------
extern "C" void kernel_launch(void* const* d_in, const int* in_sizes, int n_in,
                              void* d_out, int out_size, void* d_ws, size_t ws_size,
                              hipStream_t stream)
{
    (void)in_sizes; (void)n_in; (void)out_size; (void)ws_size;
    const float* query = (const float*)d_in[0];
    const float* key   = (const float*)d_in[1];
    const float* value = (const float*)d_in[2];
    const float* Wq = (const float*)d_in[3];
    const float* bq = (const float*)d_in[4];
    const float* Wk = (const float*)d_in[5];
    const float* bk = (const float*)d_in[6];
    const float* Wv = (const float*)d_in[7];
    const float* bv = (const float*)d_in[8];
    const float* Wo = (const float*)d_in[9];
    const float* bo = (const float*)d_in[10];
    const float* gamma = (const float*)d_in[11];
    const float* beta  = (const float*)d_in[12];

    const size_t BTD = (size_t)Bc * Tc * Dc;
    const size_t NROWS = (size_t)Bc * Hc * Tc;
    float* y_out = (float*)d_out;       // [B,T,D]
    float* attn  = y_out + BTD;         // [B,H,T,T]

    char* w = (char*)d_ws;
    _Float16* qh  = (_Float16*)w;  w += BTD * sizeof(_Float16);
    _Float16* kh  = (_Float16*)w;  w += BTD * sizeof(_Float16);
    _Float16* vhh = (_Float16*)w;  w += BTD * sizeof(_Float16);
    _Float16* xh  = (_Float16*)w;  w += BTD * sizeof(_Float16);
    float*    ybuf = (float*)w;    w += BTD * sizeof(float);
    float*    rowmax = (float*)w;  w += NROWS * sizeof(float);
    float*    rowinv = (float*)w;

    const int M = Bc * Tc;              // 4096
    dim3 blk(256);
    dim3 gproj(Dc / 64, M / 128);

    gemm_bias_kernel<float, _Float16, false><<<gproj, blk, 0, stream>>>(
        query, Wq, bq, nullptr, qh, M, Dc, Dc);
    gemm_bias_kernel<float, _Float16, false><<<gproj, blk, 0, stream>>>(
        key, Wk, bk, nullptr, kh, M, Dc, Dc);
    gemm_bias_kernel<float, _Float16, false><<<gproj, blk, 0, stream>>>(
        value, Wv, bv, nullptr, vhh, M, Dc, Dc);

    dim3 gsc(Tc / 64, Tc / 128, Bc * Hc);
    scores_kernel<<<gsc, blk, 0, stream>>>(qh, kh, attn);

    softmax_stats_kernel<<<dim3((unsigned)NROWS), blk, 0, stream>>>(
        attn, rowmax, rowinv);

    attnv_fused_kernel<<<dim3(Tc / 128, Bc * Hc), blk, 0, stream>>>(
        attn, vhh, rowmax, rowinv, xh);

    gemm_bias_kernel<_Float16, float, true><<<gproj, blk, 0, stream>>>(
        xh, Wo, bo, query, ybuf, M, Dc, Dc);

    layernorm_kernel<<<dim3(M), blk, 0, stream>>>(ybuf, gamma, beta, y_out);
}
